// Retation_58643483459803
// MI455X (gfx1250) — matrix operand order repaired and checked by reading, not compiled, not run on hardware
//
#include <hip/hip_runtime.h>
#include <hip/hip_bf16.h>
#include <stdint.h>

#define S_LEN 2048
#define D_DIM 2048
#define H_N   16
#define HD    128

typedef __attribute__((ext_vector_type(16))) __bf16 v16bf;
typedef __attribute__((ext_vector_type(8)))  float  v8f;
typedef __attribute__((ext_vector_type(4)))  float  f32x4;

union BF16x16 { v16bf v; uint4 q[2]; unsigned short s[16]; };
union F8      { v8f v; float f[8]; };

__device__ __forceinline__ unsigned short f32_to_bf16(float f) {
    unsigned int u = __float_as_uint(f);
    u += 0x7FFFu + ((u >> 16) & 1u);           // round-to-nearest-even
    return (unsigned short)(u >> 16);
}

// ---------------------------------------------------------------------------
// CDNA5 async global->LDS (ASYNCcnt) with portable fallback.
// One call = one lane's 16B slice of a wave-wide global_load_async_to_lds_b128.
// Builtin signature (from hipcc diagnostic): param0 = vector_size(16) int in
// global AS (prints as "__device__"), non-const; param1 = LDS counterpart.
// ---------------------------------------------------------------------------
#if defined(__HIP_DEVICE_COMPILE__) && defined(__gfx1250__) && \
    __has_builtin(__builtin_amdgcn_global_load_async_to_lds_b128)
#define HAS_ASYNC_LDS 1
#endif

#if defined(HAS_ASYNC_LDS)
typedef int v4i __attribute__((vector_size(16)));
typedef __attribute__((address_space(1))) v4i* gv4i_p;
typedef __attribute__((address_space(3))) v4i* lv4i_p;
#endif

__device__ __forceinline__ void async_copy16(const void* g, void* l) {
#if defined(HAS_ASYNC_LDS)
    __builtin_amdgcn_global_load_async_to_lds_b128(
        (gv4i_p)(uintptr_t)g,
        (lv4i_p)(unsigned)(uintptr_t)l,
        0, 0);
#else
    *reinterpret_cast<uint4*>(l) = *reinterpret_cast<const uint4*>(g);
#endif
}

__device__ __forceinline__ void async_wait_all() {
#if defined(HAS_ASYNC_LDS)
    asm volatile("s_wait_asynccnt 0" ::: "memory");
#endif
}

// ---------------------------------------------------------------------------
// fp32 -> bf16 bulk convert (vectorized x4, non-temporal source reads)
// ---------------------------------------------------------------------------
__global__ void cvt_f32_bf16(const float* __restrict__ src,
                             unsigned short* __restrict__ dst, int n4) {
    int i = blockIdx.x * blockDim.x + threadIdx.x;
    if (i >= n4) return;
    f32x4 f = __builtin_nontemporal_load(reinterpret_cast<const f32x4*>(src) + i);
    ushort4 o;
    o.x = f32_to_bf16(f.x); o.y = f32_to_bf16(f.y);
    o.z = f32_to_bf16(f.z); o.w = f32_to_bf16(f.w);
    reinterpret_cast<ushort4*>(dst)[i] = o;
}

// ---------------------------------------------------------------------------
// Generic bf16 WMMA GEMM:  C[m,n] = sum_k A[m,k] * W[n,k] + bias[n]
// MODE 0: Cf (fp32, MxN row-major).  MODE 1: Ct (bf16, NxM transposed).
// Each wave computes a 32(M) x 64(N) tile; block = 8 waves.
// ---------------------------------------------------------------------------
template<int MODE>
__global__ __launch_bounds__(256) void gemm_bf16_wmma(
    const unsigned short* __restrict__ A,
    const unsigned short* __restrict__ W,
    const float* __restrict__ bias,
    float* __restrict__ Cf,
    unsigned short* __restrict__ Ct,
    int M, int N, int K)
{
    const int wave = threadIdx.x >> 5;
    const int lane = threadIdx.x & 31;
    const int hi   = lane >> 4;
    const int lm   = lane & 15;
    const int m0   = blockIdx.x * 32;
    const int n0   = (blockIdx.y * 8 + wave) * 64;

    F8 acc[2][4] = {};

    for (int k0 = 0; k0 < K; k0 += 32) {
        BF16x16 a[2];
#pragma unroll
        for (int i = 0; i < 2; ++i) {
            const unsigned short* p = A + (size_t)(m0 + i*16 + lm) * K + k0 + hi*8;
            a[i].q[0] = *reinterpret_cast<const uint4*>(p);
            a[i].q[1] = *reinterpret_cast<const uint4*>(p + 16);
        }
        BF16x16 b[4];
#pragma unroll
        for (int j = 0; j < 4; ++j) {
            const unsigned short* p = W + (size_t)(n0 + j*16 + lm) * K + k0 + hi*8;
            b[j].q[0] = *reinterpret_cast<const uint4*>(p);
            b[j].q[1] = *reinterpret_cast<const uint4*>(p + 16);
        }
#pragma unroll
        for (int i = 0; i < 2; ++i)
#pragma unroll
            for (int j = 0; j < 4; ++j)
                acc[i][j].v = __builtin_amdgcn_wmma_f32_16x16x32_bf16(
                    false, a[i].v, false, b[j].v, (short)0, acc[i][j].v, false, false);
    }

#pragma unroll
    for (int j = 0; j < 4; ++j) {
        const int col = n0 + j*16 + lm;
        const float bv = bias ? bias[col] : 0.0f;
#pragma unroll
        for (int i = 0; i < 2; ++i) {
#pragma unroll
            for (int r = 0; r < 8; ++r) {
                const int row = m0 + i*16 + r + 8*hi;
                const float val = acc[i][j].f[r] + bv;
                if (MODE == 0) Cf[(size_t)row * N + col] = val;
                else           Ct[(size_t)col * M + row] = f32_to_bf16(val);
            }
        }
    }
}

// ---------------------------------------------------------------------------
// RoPE: rotate q,k pairs; fold hd^-0.5 into k; fp32 -> bf16
// ---------------------------------------------------------------------------
__global__ void rope_kernel(const float* __restrict__ qf, const float* __restrict__ kf,
                            const float* __restrict__ fcos, const float* __restrict__ fsin,
                            unsigned short* __restrict__ qb, unsigned short* __restrict__ kb) {
    int idx = blockIdx.x * blockDim.x + threadIdx.x;
    if (idx >= S_LEN * H_N * (HD / 2)) return;
    const int j = idx & (HD/2 - 1);
    const int h = (idx >> 6) & (H_N - 1);
    const int s = idx >> 10;
    const size_t base = (size_t)s * D_DIM + h * HD + 2*j;
    const float c  = fcos[s * (HD/2) + j];
    const float sn = fsin[s * (HD/2) + j];
    const float qr = __builtin_nontemporal_load(qf + base);
    const float qi = __builtin_nontemporal_load(qf + base + 1);
    qb[base]     = f32_to_bf16(qr*c - qi*sn);
    qb[base + 1] = f32_to_bf16(qr*sn + qi*c);
    const float scale = 0.08838834764831845f;   // 128^-0.5
    const float kr = __builtin_nontemporal_load(kf + base)     * scale;
    const float ki = __builtin_nontemporal_load(kf + base + 1) * scale;
    kb[base]     = f32_to_bf16(kr*c - ki*sn);
    kb[base + 1] = f32_to_bf16(kr*sn + ki*c);
}

// ---------------------------------------------------------------------------
// Fused attention, one wave per (head, 16-query strip), 8 waves/block share a
// head. k and vT chunks are double-buffered in LDS via async global->LDS so
// each chunk is fetched once per block instead of once per wave (8x less L2
// traffic). LDS rows padded (k: 136 elem, v/ret: 40 elem) for conflict-free
// ds_load_b128 fragment reads. Mask/gate are streamed non-temporally.
// grid = (H, S/16/8), block = 256
// ---------------------------------------------------------------------------
__global__ __launch_bounds__(256) void attn_wmma(
    const unsigned short* __restrict__ qb,
    const unsigned short* __restrict__ kb,
    const unsigned short* __restrict__ vT,   // [D, S] bf16 (transposed)
    const float* __restrict__ mask,          // [H, S, S]
    const float* __restrict__ gf,            // [S, D] fp32 gate
    const float* __restrict__ nw,            // [HD]
    unsigned short* __restrict__ ab)         // [S, D] bf16 out
{
    __shared__ unsigned short lds_k[2][32 * 136];   // 272B row stride (17x16B)
    __shared__ unsigned short lds_v[2][128 * 40];   // 80B row stride (5x16B)
    __shared__ unsigned short lds_ret[8][16 * 40];  // per-wave reshape tile

    const int tid  = threadIdx.x;
    const int wave = tid >> 5;
    const int lane = tid & 31;
    const int hi = lane >> 4, lm = lane & 15;
    const int head  = blockIdx.x;
    const int strip = blockIdx.y * 8 + wave;
    const int q0 = strip * 16;
    unsigned short* myt = lds_ret[wave];

    // cooperative stage of one 32-key chunk of k: 32 x 128 bf16 (8KB)
    auto stage_k = [&](int buf, int jb) {
#pragma unroll
        for (int u = 0; u < 2; ++u) {
            const int id  = tid * 2 + u;        // 512 x 16B transfers
            const int row = id >> 4;            // key 0..31
            const int c16 = id & 15;            // 16B column chunk
            const unsigned short* g = kb + (size_t)(jb + row) * D_DIM + head*HD + c16*8;
            async_copy16(g, &lds_k[buf][row*136 + c16*8]);
        }
    };
    // cooperative stage of one 32-key chunk of vT: 128 x 32 bf16 (8KB)
    auto stage_v = [&](int buf, int jb) {
#pragma unroll
        for (int u = 0; u < 2; ++u) {
            const int id  = tid * 2 + u;
            const int row = id >> 2;            // hd 0..127
            const int c4  = id & 3;
            const unsigned short* g = vT + (size_t)(head*HD + row) * S_LEN + jb + c4*8;
            async_copy16(g, &lds_v[buf][row*40 + c4*8]);
        }
    };

    // Q A-fragments: 16 x 128 as four K=32 chunks, live across both passes
    BF16x16 qa[4];
#pragma unroll
    for (int c = 0; c < 4; ++c) {
        const unsigned short* p = qb + (size_t)(q0 + lm) * D_DIM + head*HD + c*32 + hi*8;
        qa[c].q[0] = *reinterpret_cast<const uint4*>(p);
        qa[c].q[1] = *reinterpret_cast<const uint4*>(p + 16);
    }

    // ---- pass 1: L1 denominators per query row (k from LDS) ----
    float rowAbs[8] = {0,0,0,0,0,0,0,0};
    stage_k(0, 0);
    for (int kc = 0; kc < S_LEN / 32; ++kc) {
        async_wait_all();
        __syncthreads();                        // chunk kc ready; kc-1 consumed
        if (kc + 1 < S_LEN / 32) stage_k((kc + 1) & 1, (kc + 1) * 32);
        const unsigned short* kbuf = lds_k[kc & 1];
#pragma unroll
        for (int t = 0; t < 2; ++t) {
            F8 sc = {};
#pragma unroll
            for (int c = 0; c < 4; ++c) {
                BF16x16 kfr;
                const unsigned short* lp = kbuf + (t*16 + lm)*136 + c*32 + hi*8;
                kfr.q[0] = *reinterpret_cast<const uint4*>(lp);
                kfr.q[1] = *reinterpret_cast<const uint4*>(lp + 16);
                sc.v = __builtin_amdgcn_wmma_f32_16x16x32_bf16(
                    false, qa[c].v, false, kfr.v, (short)0, sc.v, false, false);
            }
#pragma unroll
            for (int r = 0; r < 8; ++r) rowAbs[r] += fabsf(sc.f[r]);
        }
    }
    float inv_denom[8];
#pragma unroll
    for (int r = 0; r < 8; ++r) {
        float t = rowAbs[r];
        t += __shfl_xor(t, 1); t += __shfl_xor(t, 2);
        t += __shfl_xor(t, 4); t += __shfl_xor(t, 8);   // within 16-lane half
        t = fminf(fmaxf(t, 1.0f), 50000.0f);
        inv_denom[r] = 1.0f / t;
    }

    // ---- pass 2: out = (sc/denom + mask) @ v (k, v from LDS) ----
    F8 out[8] = {};
    __syncthreads();                            // pass-1 buffers idle
    stage_k(0, 0);
    stage_v(0, 0);
    for (int kc = 0; kc < S_LEN / 32; ++kc) {
        async_wait_all();
        __syncthreads();
        if (kc + 1 < S_LEN / 32) {
            stage_k((kc + 1) & 1, (kc + 1) * 32);
            stage_v((kc + 1) & 1, (kc + 1) * 32);
        }
        const int jb = kc * 32;
        const unsigned short* kbuf = lds_k[kc & 1];
        const unsigned short* vbuf = lds_v[kc & 1];
#pragma unroll
        for (int t = 0; t < 2; ++t) {
            const int j0 = jb + t * 16;
            F8 sc = {};
#pragma unroll
            for (int c = 0; c < 4; ++c) {
                BF16x16 kfr;
                const unsigned short* lp = kbuf + (t*16 + lm)*136 + c*32 + hi*8;
                kfr.q[0] = *reinterpret_cast<const uint4*>(lp);
                kfr.q[1] = *reinterpret_cast<const uint4*>(lp + 16);
                sc.v = __builtin_amdgcn_wmma_f32_16x16x32_bf16(
                    false, qa[c].v, false, kfr.v, (short)0, sc.v, false, false);
            }
#pragma unroll
            for (int r = 0; r < 8; ++r) {
                const int row = q0 + r + 8*hi;
                const float mv = __builtin_nontemporal_load(
                    mask + ((size_t)head * S_LEN + row) * S_LEN + j0 + lm);
                const float val = sc.f[r] * inv_denom[r] + mv;
                myt[(r + 8*hi) * 40 + t*16 + lm] = f32_to_bf16(val);
            }
        }
        asm volatile("s_wait_dscnt 0" ::: "memory");   // ret tile visible to re-read

        BF16x16 ra;        // reshape C-layout tile -> A-fragment via LDS
        {
            const unsigned short* lp = myt + lm*40 + hi*8;
            ra.q[0] = *reinterpret_cast<const uint4*>(lp);
            ra.q[1] = *reinterpret_cast<const uint4*>(lp + 16);
        }
#pragma unroll
        for (int nt = 0; nt < 8; ++nt) {
            BF16x16 vb;
            const unsigned short* lp = vbuf + (nt*16 + lm)*40 + hi*8;
            vb.q[0] = *reinterpret_cast<const uint4*>(lp);
            vb.q[1] = *reinterpret_cast<const uint4*>(lp + 16);
            out[nt].v = __builtin_amdgcn_wmma_f32_16x16x32_bf16(
                false, ra.v, false, vb.v, (short)0, out[nt].v, false, false);
        }
    }

    // ---- epilogue: RMS norm over hd, * norm_w * gate, bf16 store ----
#pragma unroll
    for (int r = 0; r < 8; ++r) {
        float t = 0.f;
#pragma unroll
        for (int nt = 0; nt < 8; ++nt) t += out[nt].f[r] * out[nt].f[r];
        t += __shfl_xor(t, 1); t += __shfl_xor(t, 2);
        t += __shfl_xor(t, 4); t += __shfl_xor(t, 8);
        const float inv = rsqrtf(t * (1.0f / HD) + 1e-5f);
        const int row = q0 + r + 8*hi;
#pragma unroll
        for (int nt = 0; nt < 8; ++nt) {
            const int col = nt*16 + lm;
            const size_t gi = (size_t)row * D_DIM + head*HD + col;
            const float val = out[nt].f[r] * inv * nw[col] *
                              __builtin_nontemporal_load(gf + gi);
            ab[gi] = f32_to_bf16(val);
        }
    }
}

// ---------------------------------------------------------------------------
extern "C" void kernel_launch(void* const* d_in, const int* in_sizes, int n_in,
                              void* d_out, int out_size, void* d_ws, size_t ws_size,
                              hipStream_t stream)
{
    const float* x    = (const float*)d_in[0];
    const float* fcos = (const float*)d_in[1];
    const float* fsin = (const float*)d_in[2];
    const float* mask = (const float*)d_in[3];
    const float* Wq = (const float*)d_in[4];   const float* bq = (const float*)d_in[5];
    const float* Wk = (const float*)d_in[6];   const float* bk = (const float*)d_in[7];
    const float* Wv = (const float*)d_in[8];   const float* bv = (const float*)d_in[9];
    const float* Wg = (const float*)d_in[10];  const float* bg = (const float*)d_in[11];
    const float* Wp = (const float*)d_in[12];  const float* bp = (const float*)d_in[13];
    const float* nw = (const float*)d_in[14];

    const size_t SD = (size_t)S_LEN * D_DIM;   // 4M elements
    char* ws = (char*)d_ws;
    unsigned short* xb  = (unsigned short*)ws;  ws += SD * 2;
    unsigned short* wqb = (unsigned short*)ws;  ws += SD * 2;
    unsigned short* wkb = (unsigned short*)ws;  ws += SD * 2;
    unsigned short* wvb = (unsigned short*)ws;  ws += SD * 2;
    unsigned short* wgb = (unsigned short*)ws;  ws += SD * 2;
    unsigned short* wpb = (unsigned short*)ws;  ws += SD * 2;
    unsigned short* qb2 = (unsigned short*)ws;  ws += SD * 2;
    unsigned short* kb2 = (unsigned short*)ws;  ws += SD * 2;
    unsigned short* vT  = (unsigned short*)ws;  ws += SD * 2;
    unsigned short* ab  = (unsigned short*)ws;  ws += SD * 2;
    float* qf = (float*)ws;  ws += SD * 4;
    float* kf = (float*)ws;  ws += SD * 4;
    float* gf = (float*)ws;  ws += SD * 4;

    // 1) fp32 -> bf16 conversions (x + 5 weights)
    const int n4 = (int)(SD / 4);
    const dim3 cb(256), cg((n4 + 255) / 256);
    cvt_f32_bf16<<<cg, cb, 0, stream>>>(x,  xb,  n4);
    cvt_f32_bf16<<<cg, cb, 0, stream>>>(Wq, wqb, n4);
    cvt_f32_bf16<<<cg, cb, 0, stream>>>(Wk, wkb, n4);
    cvt_f32_bf16<<<cg, cb, 0, stream>>>(Wv, wvb, n4);
    cvt_f32_bf16<<<cg, cb, 0, stream>>>(Wg, wgb, n4);
    cvt_f32_bf16<<<cg, cb, 0, stream>>>(Wp, wpb, n4);

    // 2) input projections (WMMA bf16)
    const dim3 gg(D_DIM / 32, D_DIM / (64 * 8)), gb(256);
    gemm_bf16_wmma<0><<<gg, gb, 0, stream>>>(xb, wqb, bq, qf, nullptr, S_LEN, D_DIM, D_DIM);
    gemm_bf16_wmma<0><<<gg, gb, 0, stream>>>(xb, wkb, bk, kf, nullptr, S_LEN, D_DIM, D_DIM);
    gemm_bf16_wmma<1><<<gg, gb, 0, stream>>>(xb, wvb, bv, nullptr, vT,  S_LEN, D_DIM, D_DIM);
    gemm_bf16_wmma<0><<<gg, gb, 0, stream>>>(xb, wgb, bg, gf, nullptr, S_LEN, D_DIM, D_DIM);

    // 3) RoPE (+ k scaling), fp32 -> bf16
    const int nrope = S_LEN * H_N * (HD / 2);
    rope_kernel<<<(nrope + 255) / 256, 256, 0, stream>>>(qf, kf, fcos, fsin, qb2, kb2);

    // 4) fused attention (denominator + AV + RMS + gate)
    attn_wmma<<<dim3(H_N, S_LEN / 16 / 8), 256, 0, stream>>>(qb2, kb2, vT, mask, gf, nw, ab);

    // 5) output projection -> fp32 d_out
    gemm_bf16_wmma<0><<<gg, gb, 0, stream>>>(ab, wpb, bp, (float*)d_out, nullptr, S_LEN, D_DIM, D_DIM);
}
